// ConvDecoder_23278722744322
// MI455X (gfx1250) — matrix operand order repaired and checked by reading, not compiled, not run on hardware
//
#include <hip/hip_runtime.h>
#include <hip/hip_bf16.h>
#include <stdint.h>

// ---------------------------------------------------------------------------
// Problem constants (from reference): B=4, N_TGT=4096, N_GRID=8192, C_CONV=64,
// C_OUT=2, D=1.  x_context / y_context are unused by the reference.
// ---------------------------------------------------------------------------
#define BB      4
#define TT      4096
#define GG      8192
#define CC      64
#define NCHUNK  (GG / 32)         // 256 K-chunks of 32
#define KSPLIT  4
#define JPS     (NCHUNK / KSPLIT) // 64 chunks per split
#define OUT_ELEMS (BB * TT * 2)   // 32768

typedef __attribute__((ext_vector_type(16))) _Float16 v16h;
typedef __attribute__((ext_vector_type(8)))  float    v8f;

// ---------------------------------------------------------------------------
// Kernel 1: pack r (f32, [b][c][g]) into FP16 WMMA-B-operand order:
//   rp[(((b*NCHUNK + j)*4 + tile)*32 + lane)*16 + e]
//     = (f16) r[b][16*tile + (lane&15)][32*j + 16*(lane>>4) + e]
// so the main kernel's B-tile loads are 32 contiguous bytes per lane.
// ---------------------------------------------------------------------------
__global__ void pack_r_f16(const float* __restrict__ r, _Float16* __restrict__ rp) {
    int idx = blockIdx.x * 256 + threadIdx.x;       // < BB*CC*GG = 2,097,152
    int e    =  idx        & 15;
    int lane = (idx >> 4)  & 31;
    int tile = (idx >> 9)  & 3;
    int j    = (idx >> 11) & 255;
    int b    =  idx >> 19;
    int c = tile * 16 + (lane & 15);
    int g = j * 32 + (lane >> 4) * 16 + e;
    rp[idx] = (_Float16)r[((size_t)(b * CC + c)) * GG + g];
}

// ---------------------------------------------------------------------------
// Helpers for the fused kernel
// ---------------------------------------------------------------------------
__device__ __forceinline__ void load_btiles(const _Float16* __restrict__ base,
                                            v16h& t0, v16h& t1, v16h& t2, v16h& t3) {
    t0 = *(const v16h*)(base);
    t1 = *(const v16h*)(base + 512);
    t2 = *(const v16h*)(base + 1024);
    t3 = *(const v16h*)(base + 1536);
}

// Build the two 16x32 FP16 A tiles (rows m and m+16) for one K-chunk.
// A-layout K-slots for a lane: [16h, 16h+16); element e of the
// v16h maps linearly onto those 16 grid points.
//
// The -0.5*log2(e) exponent constant is pre-folded into the distance scale:
// with invsk = exp(-sigma)*sqrt(0.5*log2 e) and xtsk = x_t*invsk,
//   rbf = exp2( -(x_g*invsk - xtsk)^2 )
// -> 3 VALU ops per element (fma, sqr, exp with folded negate).
__device__ __forceinline__ void rbf_pair(const float* __restrict__ x_grid, int g0,
                                         int h, float invsk, float xts0, float xts1,
                                         v16h& a0, v16h& a1) {
    const float4 xa = *(const float4*)(x_grid + g0 + 16 * h);
    const float4 xb = *(const float4*)(x_grid + g0 + 16 * h + 4);
    const float4 xc = *(const float4*)(x_grid + g0 + 16 * h + 8);
    const float4 xd = *(const float4*)(x_grid + g0 + 16 * h + 12);
    const float xs[16] = { xa.x, xa.y, xa.z, xa.w,  xb.x, xb.y, xb.z, xb.w,
                           xc.x, xc.y, xc.z, xc.w,  xd.x, xd.y, xd.z, xd.w };
#pragma unroll
    for (int e = 0; e < 16; ++e) {
        float d0 = __builtin_fmaf(xs[e], invsk, -xts0);
        float d1 = __builtin_fmaf(xs[e], invsk, -xts1);
        a0[e] = (_Float16)__builtin_amdgcn_exp2f(-(d0 * d0));
        a1[e] = (_Float16)__builtin_amdgcn_exp2f(-(d1 * d1));
    }
}

#define WMMA8(A0, A1, T0, T1, T2, T3)                                                        \
    acc00 = __builtin_amdgcn_wmma_f32_16x16x32_f16(false, A0, false, T0, (short)0, acc00, false, false); \
    acc10 = __builtin_amdgcn_wmma_f32_16x16x32_f16(false, A1, false, T0, (short)0, acc10, false, false); \
    acc01 = __builtin_amdgcn_wmma_f32_16x16x32_f16(false, A0, false, T1, (short)0, acc01, false, false); \
    acc11 = __builtin_amdgcn_wmma_f32_16x16x32_f16(false, A1, false, T1, (short)0, acc11, false, false); \
    acc02 = __builtin_amdgcn_wmma_f32_16x16x32_f16(false, A0, false, T2, (short)0, acc02, false, false); \
    acc12 = __builtin_amdgcn_wmma_f32_16x16x32_f16(false, A1, false, T2, (short)0, acc12, false, false); \
    acc03 = __builtin_amdgcn_wmma_f32_16x16x32_f16(false, A0, false, T3, (short)0, acc03, false, false); \
    acc13 = __builtin_amdgcn_wmma_f32_16x16x32_f16(false, A1, false, T3, (short)0, acc13, false, false)

// ---------------------------------------------------------------------------
// Kernel 2: fused RBF + batched GEMM on the WMMA pipe.
//   grid = (TT/128, KSPLIT, BB), block = 128 (4 waves).
//   Wave w handles target rows [tb*128 + 32w, +32) x all 64 channels (two A
//   tiles against four B tiles = 8 WMMAs per K-chunk), over grid range
//   [s*2048, (s+1)*2048), with a 2-deep software pipeline on the B tiles.
// ---------------------------------------------------------------------------
__global__ __launch_bounds__(128) void fused_rbf_wmma(
    const float*    __restrict__ x_target,  // (B, T, 1)
    const float*    __restrict__ x_grid,    // (G, 1)
    const float*    __restrict__ sigma,     // (1,)
    const float*    __restrict__ Wf,        // (64, 2)
    const _Float16* __restrict__ rp,        // packed B operands
    float*          __restrict__ part)      // (KSPLIT, B, T, 2)
{
    const int lane = threadIdx.x & 31;
    const int wv   = threadIdx.x >> 5;     // wave in block: 0..3
    const int tb   = blockIdx.x;           // target block: 0..31
    const int s    = blockIdx.y;           // K split: 0..3
    const int b    = blockIdx.z;           // batch: 0..3
    const int ml   = lane & 15;            // M row (A) / N col (B,C)
    const int h    = lane >> 4;            // lane-half selector

    // invsk = exp(-sigma) * sqrt(0.5*log2 e)   (exponent constant pre-folded)
    const float SQRT_HALF_LOG2E = 0.84932180028801904f;   // sqrt(0.5/ln 2)
    const float invsk = __builtin_amdgcn_exp2f(-sigma[0] * 1.44269504088896340736f)
                        * SQRT_HALF_LOG2E;
    const int   trow = tb * 128 + wv * 32 + ml;
    const float xts0 = x_target[b * TT + trow]      * invsk;  // rows m
    const float xts1 = x_target[b * TT + trow + 16] * invsk;  // rows m+16

    v8f acc00 = {}, acc01 = {}, acc02 = {}, acc03 = {};
    v8f acc10 = {}, acc11 = {}, acc12 = {}, acc13 = {};

    const int j0 = s * JPS;
    const _Float16* rpB = rp + ((size_t)(b * NCHUNK + j0) * 4) * 32 * 16
                             + (size_t)lane * 16;

    // ---- 2-deep software pipeline over K-chunks --------------------------
    v16h p0, p1, p2, p3;   // stage A tiles
    v16h q0, q1, q2, q3;   // stage B tiles
    v16h a0, a1;

    load_btiles(rpB, p0, p1, p2, p3);                        // chunk jj=0

    for (int jj = 0; jj < JPS; jj += 2) {
        __builtin_prefetch((const void*)(rpB + (size_t)(jj + 12) * 2048), 0, 1);

        // stage 1: loads for chunk jj+1 fly under the 8 WMMAs of chunk jj
        load_btiles(rpB + (size_t)(jj + 1) * 2048, q0, q1, q2, q3);
        rbf_pair(x_grid, (j0 + jj) * 32, h, invsk, xts0, xts1, a0, a1);
        WMMA8(a0, a1, p0, p1, p2, p3);

        // stage 2: loads for chunk jj+2 fly under the 8 WMMAs of chunk jj+1
        const int jn = (jj + 2 < JPS) ? (jj + 2) : 0;        // clamp (stay in ws)
        load_btiles(rpB + (size_t)jn * 2048, p0, p1, p2, p3);
        rbf_pair(x_grid, (j0 + jj + 1) * 32, h, invsk, xts0, xts1, a0, a1);
        WMMA8(a0, a1, q0, q1, q2, q3);
    }

    // ---- Epilogue: apply W (64x2) and reduce the N dimension -------------
    // C layout: lane -> N = ml (per tile), VGPR v -> M = 8h+v.
    float w0[4], w1[4];
#pragma unroll
    for (int t = 0; t < 4; ++t) {
        w0[t] = Wf[(ml + 16 * t) * 2 + 0];
        w1[t] = Wf[(ml + 16 * t) * 2 + 1];
    }
#pragma unroll
    for (int a = 0; a < 2; ++a) {
        const v8f& c0 = a ? acc10 : acc00;
        const v8f& c1 = a ? acc11 : acc01;
        const v8f& c2 = a ? acc12 : acc02;
        const v8f& c3 = a ? acc13 : acc03;
#pragma unroll
        for (int v = 0; v < 8; ++v) {
            float p0r = c0[v] * w0[0] + c1[v] * w0[1] + c2[v] * w0[2] + c3[v] * w0[3];
            float p1r = c0[v] * w1[0] + c1[v] * w1[1] + c2[v] * w1[2] + c3[v] * w1[3];
#pragma unroll
            for (int m = 1; m < 16; m <<= 1) {
                p0r += __shfl_xor(p0r, m, 32);
                p1r += __shfl_xor(p1r, m, 32);
            }
            if (ml == 0) {
                const int row = tb * 128 + wv * 32 + 16 * a + 8 * h + v;
                float* o = part + (((size_t)s * BB + b) * TT + row) * 2;
                o[0] = p0r;
                o[1] = p1r;
            }
        }
    }
}

// ---------------------------------------------------------------------------
// Kernel 3: deterministic fixed-order reduction of the KSPLIT partials + bias.
// ---------------------------------------------------------------------------
__global__ void reduce_splits(const float* __restrict__ part,
                              const float* __restrict__ b_lin,
                              float* __restrict__ out) {
    int idx = blockIdx.x * 256 + threadIdx.x;   // < OUT_ELEMS
    float v = b_lin[idx & 1];
#pragma unroll
    for (int s = 0; s < KSPLIT; ++s)
        v += part[(size_t)s * OUT_ELEMS + idx];
    out[idx] = v;
}

// ---------------------------------------------------------------------------
// Host launcher.  Inputs (setup_inputs order):
//   0: r (B,C,G) f32   1: x_context (unused)   2: y_context (unused)
//   3: x_target (B,T,1) 4: x_grid (G,1) 5: sigma (1,) 6: W (64,2) 7: b_lin (2,)
// ws layout: [0, 4 MiB) packed f16 r;  [4 MiB, +512 KiB) split partials.
// ---------------------------------------------------------------------------
extern "C" void kernel_launch(void* const* d_in, const int* in_sizes, int n_in,
                              void* d_out, int out_size, void* d_ws, size_t ws_size,
                              hipStream_t stream) {
    (void)in_sizes; (void)n_in; (void)out_size; (void)ws_size;
    const float* r        = (const float*)d_in[0];
    const float* x_target = (const float*)d_in[3];
    const float* x_grid   = (const float*)d_in[4];
    const float* sigma    = (const float*)d_in[5];
    const float* Wf       = (const float*)d_in[6];
    const float* b_lin    = (const float*)d_in[7];
    float* out = (float*)d_out;

    _Float16* rp  = (_Float16*)d_ws;
    float*    prt = (float*)((char*)d_ws + (size_t)BB * CC * GG * sizeof(_Float16));

    // 1) pack r -> f16 WMMA-B order (2,097,152 elems)
    pack_r_f16<<<(BB * CC * GG) / 256, 256, 0, stream>>>(r, rp);

    // 2) fused RBF + GEMM + W epilogue
    fused_rbf_wmma<<<dim3(TT / 128, KSPLIT, BB), 128, 0, stream>>>(
        x_target, x_grid, sigma, Wf, rp, prt);

    // 3) deterministic split reduction + bias
    reduce_splits<<<OUT_ELEMS / 256, 256, 0, stream>>>(prt, b_lin, out);
}
